// TinyRecursiveModel_41480794145094
// MI455X (gfx1250) — matrix operand order repaired and checked by reading, not compiled, not run on hardware
//
#include <hip/hip_runtime.h>

#define NB 65536
#define IN_F 200
#define NLAYERS 2
#define NSTEPS 16
#define LN_EPS 1e-5f
#define SROW 80   // padded f16 row stride in LDS (160B, 16B-aligned -> ds_read_b128)

typedef _Float16 v2h  __attribute__((ext_vector_type(2)));
typedef _Float16 v8h  __attribute__((ext_vector_type(8)));
typedef _Float16 v16h __attribute__((ext_vector_type(16)));
typedef float    v8f  __attribute__((ext_vector_type(8)));
typedef int      v8i  __attribute__((ext_vector_type(8)));

__device__ __forceinline__ int pkf16(float lo, float hi) {
  v2h p; p[0] = (_Float16)lo; p[1] = (_Float16)hi;
  return __builtin_bit_cast(int, p);
}

// A fragment (16x32 f16, M=weight row): lanes 0-15 hold K {k..k+15},
// lanes 16-31 hold K {k+16..k+31}.  Two aligned 16B LDS loads.
__device__ __forceinline__ v16h ld_afrag(const _Float16* rowp, int kbase, int half) {
  const int k0 = kbase + half * 16;
  v8h lo = *(const v8h*)(rowp + k0);
  v8h hi = *(const v8h*)(rowp + k0 + 8);
  return __builtin_shufflevector(lo, hi, 0, 1, 2, 3, 4, 5, 6, 7,
                                 8, 9, 10, 11, 12, 13, 14, 15);
}

// State lives transposed in registers: lane (half,n), tile t, elem r holds
// feature m = t*16 + half*8 + r of batch row n.  Build the two B fragments
// (K = 0..31 and 32..63) for the next W x h^T WMMA purely in registers:
// B-layout lane (half,n) dword d = feature pair {kb + half*16 + 2d, +1} of row n.
__device__ __forceinline__ void pack_state(const v8f h[4], int half, v16h* b0, v16h* b1) {
  int P[4][4], S[4][4];
#pragma unroll
  for (int t = 0; t < 4; t++)
#pragma unroll
    for (int j = 0; j < 4; j++)
      P[t][j] = pkf16(h[t][2 * j], h[t][2 * j + 1]);   // features t*16+half*8+{2j,2j+1}
#pragma unroll
  for (int t = 0; t < 4; t++)
#pragma unroll
    for (int j = 0; j < 4; j++)
      S[t][j] = __shfl_xor(P[t][j], 16, 32);           // partner half's pairs
  v8i i0, i1;
#pragma unroll
  for (int j = 0; j < 4; j++) {
    i0[j]     = half ? S[1][j] : P[0][j];   // K 0..7   | 16..23
    i0[j + 4] = half ? P[1][j] : S[0][j];   // K 8..15  | 24..31
    i1[j]     = half ? S[3][j] : P[2][j];   // K 32..39 | 48..55
    i1[j + 4] = half ? P[3][j] : S[2][j];   // K 40..47 | 56..63
  }
  *b0 = __builtin_bit_cast(v16h, i0);
  *b1 = __builtin_bit_cast(v16h, i1);
}

// out = W x hin^T + bias : A = 64x64 weights (LDS), B = state (registers).
// hin is fully consumed by pack_state before out is written (out may alias hin).
__device__ __forceinline__ void matmulT(const v8f hin[4], const _Float16* w,
                                        const float* bias64, int n, int half, v8f out[4]) {
  v16h b0, b1;
  pack_state(hin, half, &b0, &b1);
#pragma unroll
  for (int t = 0; t < 4; t++) {
    v8f c = *(const v8f*)(bias64 + t * 16 + half * 8);  // bias straight into acc
    const _Float16* wr = w + (t * 16 + n) * SROW;
    v16h a0 = ld_afrag(wr, 0, half);
    c = __builtin_amdgcn_wmma_f32_16x16x32_f16(false, a0, false, b0, (short)0, c, false, false);
    v16h a1 = ld_afrag(wr, 32, half);
    c = __builtin_amdgcn_wmma_f32_16x16x32_f16(false, a1, false, b1, (short)0, c, false, false);
    out[t] = c;
  }
}

// Row n is split across the lane pair (n, n+16): local 32-sum + one xor-16 shuffle.
__device__ __forceinline__ void lnormT(v8f acc[4], const float* g64, const float* b64, int half) {
  float s = 0.0f, q = 0.0f;
#pragma unroll
  for (int t = 0; t < 4; t++)
#pragma unroll
    for (int r = 0; r < 8; r++) { float v = acc[t][r]; s += v; q += v * v; }
  s += __shfl_xor(s, 16, 32);
  q += __shfl_xor(q, 16, 32);
  float mean = s * (1.0f / 64.0f);
  float var  = q * (1.0f / 64.0f) - mean * mean;
  float rs   = rsqrtf(var + LN_EPS);
#pragma unroll
  for (int t = 0; t < 4; t++) {
    v8f g  = *(const v8f*)(g64 + t * 16 + half * 8);
    v8f bb = *(const v8f*)(b64 + t * 16 + half * 8);
#pragma unroll
    for (int r = 0; r < 8; r++)
      acc[t][r] = (acc[t][r] - mean) * rs * g[r] + bb[r];
  }
}

__global__ __launch_bounds__(256, 4) void trm_fused_kernel(
    const float* __restrict__ x, const float* __restrict__ input_w, const float* __restrict__ input_b,
    const float* __restrict__ Wqkv, const float* __restrict__ bqkv,
    const float* __restrict__ Wo, const float* __restrict__ bo,
    const float* __restrict__ ln1_g, const float* __restrict__ ln1_b,
    const float* __restrict__ W1, const float* __restrict__ b1,
    const float* __restrict__ W2, const float* __restrict__ b2,
    const float* __restrict__ ln2_g, const float* __restrict__ ln2_b,
    const float* __restrict__ out_w, const float* __restrict__ out_b,
    float* __restrict__ out) {
  __shared__ __attribute__((aligned(16))) _Float16 sh_w[NLAYERS][4][64][SROW]; // Wv,Wo,W1,W2
  __shared__ __attribute__((aligned(32))) float sh_bias[NLAYERS][4][64];       // bv,bo,b1,b2
  __shared__ __attribute__((aligned(32))) float sh_ln[NLAYERS][4][64];         // g1,b1,g2,b2

  const int tid = threadIdx.x;

  // ---- cooperative weight load: f32 -> f16 into LDS (once per block) ----
#pragma unroll 1
  for (int idx = tid; idx < NLAYERS * 4 * 64 * 64; idx += 256) {
    int l = idx >> 14;
    int rem = idx & 16383;
    int m = rem >> 12;
    int rc = rem & 4095;
    int row = rc >> 6, col = rc & 63;
    const float* src;
    if (m == 0)      src = Wqkv + (l * 192 + 128) * 64;  // Wv slice
    else if (m == 1) src = Wo + l * 64 * 64;
    else if (m == 2) src = W1 + l * 64 * 64;
    else             src = W2 + l * 64 * 64;
    sh_w[l][m][row][col] = (_Float16)src[row * 64 + col];
  }
#pragma unroll 1
  for (int idx = tid; idx < NLAYERS * 4 * 64; idx += 256) {
    int l = idx >> 8; int m = (idx >> 6) & 3; int j = idx & 63;
    const float* bsrc;
    if (m == 0)      bsrc = bqkv + l * 192 + 128;
    else if (m == 1) bsrc = bo + l * 64;
    else if (m == 2) bsrc = b1 + l * 64;
    else             bsrc = b2 + l * 64;
    sh_bias[l][m][j] = bsrc[j];
    const float* lsrc;
    if (m == 0)      lsrc = ln1_g + l * 64;
    else if (m == 1) lsrc = ln1_b + l * 64;
    else if (m == 2) lsrc = ln2_g + l * 64;
    else             lsrc = ln2_b + l * 64;
    sh_ln[l][m][j] = lsrc[j];
  }
  __syncthreads();

  const int lane = tid & 31, half = lane >> 4, n = lane & 15;
  const int rowBase = blockIdx.x * 128 + (tid >> 5) * 16;

  // ---- input projection (transposed): h^T = input_w x x^T + input_b ----
  v8f hacc[4];
#pragma unroll
  for (int t = 0; t < 4; t++) {
    const float* bp = input_b + t * 16 + half * 8;
    float4 b0 = *(const float4*)bp;
    float4 b1v = *(const float4*)(bp + 4);
    hacc[t] = (v8f){b0.x, b0.y, b0.z, b0.w, b1v.x, b1v.y, b1v.z, b1v.w};
  }
  const float* xrow = x + (size_t)(rowBase + n) * IN_F;
  // K = 0..191: rolled loop, everything in-bounds (no guards needed)
#pragma unroll 1
  for (int kc = 0; kc < 6; kc++) {
    const float* xp = xrow + kc * 32 + half * 16;
    float4 f0 = *(const float4*)(xp);
    float4 f1 = *(const float4*)(xp + 4);
    float4 f2 = *(const float4*)(xp + 8);
    float4 f3 = *(const float4*)(xp + 12);
    v8i bi;
    bi[0] = pkf16(f0.x, f0.y); bi[1] = pkf16(f0.z, f0.w);
    bi[2] = pkf16(f1.x, f1.y); bi[3] = pkf16(f1.z, f1.w);
    bi[4] = pkf16(f2.x, f2.y); bi[5] = pkf16(f2.z, f2.w);
    bi[6] = pkf16(f3.x, f3.y); bi[7] = pkf16(f3.z, f3.w);
    v16h bf = __builtin_bit_cast(v16h, bi);
#pragma unroll
    for (int t = 0; t < 4; t++) {
      const float* wr = input_w + (size_t)(t * 16 + n) * IN_F + kc * 32 + half * 16;
      float4 g0 = *(const float4*)(wr);
      float4 g1 = *(const float4*)(wr + 4);
      float4 g2 = *(const float4*)(wr + 8);
      float4 g3 = *(const float4*)(wr + 12);
      v16h a;
      a[0]=(_Float16)g0.x;  a[1]=(_Float16)g0.y;  a[2]=(_Float16)g0.z;  a[3]=(_Float16)g0.w;
      a[4]=(_Float16)g1.x;  a[5]=(_Float16)g1.y;  a[6]=(_Float16)g1.z;  a[7]=(_Float16)g1.w;
      a[8]=(_Float16)g2.x;  a[9]=(_Float16)g2.y;  a[10]=(_Float16)g2.z; a[11]=(_Float16)g2.w;
      a[12]=(_Float16)g3.x; a[13]=(_Float16)g3.y; a[14]=(_Float16)g3.z; a[15]=(_Float16)g3.w;
      hacc[t] = __builtin_amdgcn_wmma_f32_16x16x32_f16(false, a, false, bf, (short)0,
                                                       hacc[t], false, false);
    }
  }
  // K = 192..223 tail: only half==0 lanes carry K 192..199; rest zero-padded
  {
    v8i bi;
#pragma unroll
    for (int j = 0; j < 8; j++) bi[j] = 0;
    if (half == 0) {
      float4 f0 = *(const float4*)(xrow + 192);
      float4 f1 = *(const float4*)(xrow + 196);
      bi[0] = pkf16(f0.x, f0.y); bi[1] = pkf16(f0.z, f0.w);
      bi[2] = pkf16(f1.x, f1.y); bi[3] = pkf16(f1.z, f1.w);
    }
    v16h bf = __builtin_bit_cast(v16h, bi);
#pragma unroll
    for (int t = 0; t < 4; t++) {
      v16h a;
#pragma unroll
      for (int j = 0; j < 16; j++) a[j] = (_Float16)0.0f;
      if (half == 0) {
        const float* wr = input_w + (size_t)(t * 16 + n) * IN_F;
        float4 g0 = *(const float4*)(wr + 192);
        float4 g1 = *(const float4*)(wr + 196);
        a[0]=(_Float16)g0.x; a[1]=(_Float16)g0.y; a[2]=(_Float16)g0.z; a[3]=(_Float16)g0.w;
        a[4]=(_Float16)g1.x; a[5]=(_Float16)g1.y; a[6]=(_Float16)g1.z; a[7]=(_Float16)g1.w;
      }
      hacc[t] = __builtin_amdgcn_wmma_f32_16x16x32_f16(false, a, false, bf, (short)0,
                                                       hacc[t], false, false);
    }
  }

  // ---- recursive core: 16 steps x 2 layers; only hacc + one temp live ----
  v8f t1[4];
#pragma unroll 1
  for (int s = 0; s < NSTEPS; s++) {
#pragma unroll
    for (int l = 0; l < NLAYERS; l++) {
      // v = Wv x h^T + bv
      matmulT(hacc, &sh_w[l][0][0][0], &sh_bias[l][0][0], n, half, t1);
      // sa = Wo x v^T + bo  (in-place: v consumed before writes)
      matmulT(t1, &sh_w[l][1][0][0], &sh_bias[l][1][0], n, half, t1);
#pragma unroll
      for (int t = 0; t < 4; t++) hacc[t] += t1[t];
      lnormT(hacc, &sh_ln[l][0][0], &sh_ln[l][1][0], half);
      // ff1 = relu(W1 x h^T + b1)
      matmulT(hacc, &sh_w[l][2][0][0], &sh_bias[l][2][0], n, half, t1);
#pragma unroll
      for (int t = 0; t < 4; t++)
#pragma unroll
        for (int r = 0; r < 8; r++) t1[t][r] = fmaxf(t1[t][r], 0.0f);
      // ff = W2 x ff1^T + b2  (in-place)
      matmulT(t1, &sh_w[l][3][0][0], &sh_bias[l][3][0], n, half, t1);
#pragma unroll
      for (int t = 0; t < 4; t++) hacc[t] += t1[t];
      lnormT(hacc, &sh_ln[l][2][0], &sh_ln[l][3][0], half);
    }
  }

  // ---- output head: out[row] = sum_m h[m][row]*out_w[m] + out_b ----
  float p = 0.0f;
#pragma unroll
  for (int t = 0; t < 4; t++) {
    const float* op = out_w + t * 16 + half * 8;
    float4 o0 = *(const float4*)op;
    float4 o1 = *(const float4*)(op + 4);
    p += hacc[t][0] * o0.x + hacc[t][1] * o0.y + hacc[t][2] * o0.z + hacc[t][3] * o0.w;
    p += hacc[t][4] * o1.x + hacc[t][5] * o1.y + hacc[t][6] * o1.z + hacc[t][7] * o1.w;
  }
  p += __shfl_xor(p, 16, 32);
  if (half == 0) out[rowBase + n] = p + out_b[0];
}

extern "C" void kernel_launch(void* const* d_in, const int* in_sizes, int n_in,
                              void* d_out, int out_size, void* d_ws, size_t ws_size,
                              hipStream_t stream) {
  (void)in_sizes; (void)n_in; (void)d_ws; (void)ws_size; (void)out_size;
  const float* x       = (const float*)d_in[0];
  const float* input_w = (const float*)d_in[1];
  const float* input_b = (const float*)d_in[2];
  const float* Wqkv    = (const float*)d_in[3];
  const float* bqkv    = (const float*)d_in[4];
  const float* Wo      = (const float*)d_in[5];
  const float* bo      = (const float*)d_in[6];
  const float* ln1_g   = (const float*)d_in[7];
  const float* ln1_b   = (const float*)d_in[8];
  const float* W1      = (const float*)d_in[9];
  const float* b1      = (const float*)d_in[10];
  const float* W2      = (const float*)d_in[11];
  const float* b2      = (const float*)d_in[12];
  const float* ln2_g   = (const float*)d_in[13];
  const float* ln2_b   = (const float*)d_in[14];
  const float* out_w   = (const float*)d_in[15];
  const float* out_b   = (const float*)d_in[16];
  float* out = (float*)d_out;

  dim3 grid(NB / 128), block(256);
  trm_fused_kernel<<<grid, block, 0, stream>>>(x, input_w, input_b, Wqkv, bqkv, Wo, bo,
                                               ln1_g, ln1_b, W1, b1, W2, b2, ln2_g, ln2_b,
                                               out_w, out_b, out);
}
